// ConvNode_2439541424851
// MI455X (gfx1250) — matrix operand order repaired and checked by reading, not compiled, not run on hardware
//
#include <hip/hip_runtime.h>

// ---------------------------------------------------------------------------
// TBCNN tree-conv for MI455X (gfx1250, wave32, WMMA).
//   out[b,t,o] = leaky_relu( sum_{k,f} mixed[b,t,k,f] * W_k[f,o] + conv[o] )
//   mixed[k,f] = sum_{y=0..16} tree[y,f] * coef[y,k]
// GEMM done with V_WMMA_F32_16X16X32_BF16 using fp32 -> bf16 hi/lo splitting
// (Ah*Bh + Al*Bh + Ah*Bl) for near-fp32 accuracy at bf16 matrix throughput.
// ---------------------------------------------------------------------------

typedef __bf16 bf16_t;
typedef bf16_t v8bf  __attribute__((ext_vector_type(8)));
typedef bf16_t v16bf __attribute__((ext_vector_type(16)));
typedef float  v8f   __attribute__((ext_vector_type(8)));

#define Bdim   32
#define Tdim   512
#define Cdim   16
#define Fdim   256
#define Odim   256
#define Kdim   768            // 3 * Fdim
#define ROWS_PER_WG 32
#define WPAD   776            // 768 + 8 bf16 padding -> breaks LDS bank conflicts

// ---------------------------------------------------------------------------
// Prep: weights [w_t, w_r, w_l] (fp32, [F][O] each) -> transposed bf16 hi/lo
//       Wt[o][kk], kk = k*F + f.  (B-operand friendly: K contiguous per row.)
// ---------------------------------------------------------------------------
__global__ __launch_bounds__(256) void prep_weights_kernel(
    const float* __restrict__ w_t, const float* __restrict__ w_l,
    const float* __restrict__ w_r,
    bf16_t* __restrict__ Wh, bf16_t* __restrict__ Wl)
{
  int idx = blockIdx.x * 256 + threadIdx.x;          // over 3*F*O
  if (idx >= 3 * Fdim * Odim) return;
  int k   = idx / (Fdim * Odim);
  int rem = idx - k * (Fdim * Odim);
  int f   = rem >> 8;
  int o   = rem & 255;
  // weight stack order in reference is [w_t, w_r, w_l]
  const float* w = (k == 0) ? w_t : ((k == 1) ? w_r : w_l);
  float  x = w[f * Odim + o];
  bf16_t h = (bf16_t)x;
  bf16_t l = (bf16_t)(x - (float)h);
  int kk = k * Fdim + f;
  Wh[(size_t)o * Kdim + kk] = h;
  Wl[(size_t)o * Kdim + kk] = l;
}

// A operand (16-bit, 16x32): lanes 0-15 hold K[0..7] then K[16..23];
// lanes 16-31 hold K[8..15] then K[24..31]. p points at K = kk0 + koffA.
__device__ inline v16bf load_a16(const bf16_t* p) {
  v8bf c0 = *(const v8bf*)p;          // K[koffA .. koffA+7]
  v8bf c1 = *(const v8bf*)(p + 16);   // K[16+koffA .. 16+koffA+7]
  return __builtin_shufflevector(c0, c1,
      0,1,2,3,4,5,6,7,8,9,10,11,12,13,14,15);
}
// B operand (16-bit, 32x16): lanes 0-15 hold K[0..7] then K[16..23];
// lanes 16-31 hold K[8..15] then K[24..31]. p points at K = kk0 + half*8.
__device__ inline v16bf load_b16(const bf16_t* p) {
  v8bf c0 = *(const v8bf*)p;
  v8bf c1 = *(const v8bf*)(p + 16);
  return __builtin_shufflevector(c0, c1,
      0,1,2,3,4,5,6,7,8,9,10,11,12,13,14,15);
}

// ---------------------------------------------------------------------------
// Main kernel: one WG (256 thr, 8 waves) handles 32 consecutive nodes.
// ---------------------------------------------------------------------------
__global__ __launch_bounds__(256) void tree_conv_wmma_kernel(
    const float* __restrict__ nodes,      // [B*T, F]
    const float* __restrict__ cvec,       // [B*T, C, F]
    const float* __restrict__ convb,      // [O]
    const int*   __restrict__ children,   // [B*T, C]
    const bf16_t* __restrict__ Wh,        // [O, 768] bf16 hi
    const bf16_t* __restrict__ Wl,        // [O, 768] bf16 lo
    float* __restrict__ out)              // [B*T, O]
{
  __shared__ float coef[ROWS_PER_WG][17][3];
  __shared__ __align__(16) bf16_t Ah[ROWS_PER_WG][WPAD];
  __shared__ __align__(16) bf16_t Al[ROWS_PER_WG][WPAD];

  const int tid     = threadIdx.x;
  const int rowBase = blockIdx.x * ROWS_PER_WG;   // flattened b*T + t

  // ---- coefficients: one thread per node row ----
  if (tid < ROWS_PER_WG) {
    const int rg = rowBase + tid;
    float m[17];
    m[0] = 0.f;
    int nsi = 0;
    #pragma unroll
    for (int c = 0; c < Cdim; ++c) {
      int ch = children[rg * Cdim + c];
      int nz = (ch != 0) ? 1 : 0;
      nsi += nz;
      m[c + 1] = (float)nz;
    }
    const float ns = (float)nsi;
    #pragma unroll
    for (int y = 0; y < 17; ++y) {
      float ct = (y == 0) ? 1.f : 0.f;
      float cr;
      if (nsi == 1)
        cr = (y == 1) ? 0.5f : 0.f;       // single-child special case
      else
        cr = (1.f - ct) * ((float)(y - 1)) * m[y] / (ns - 1.f); // nsi==0 -> m==0 -> 0
      float cl = (1.f - ct) * (1.f - cr) * m[y];
      coef[tid][y][0] = ct;
      coef[tid][y][1] = cr;
      coef[tid][y][2] = cl;
    }
  }
  __syncthreads();

  // ---- stage 1: stream tree rows, build mixed[32][768] as bf16 hi/lo ----
  const int rsub = tid >> 6;   // 0..3  (uniform within a wave)
  const int fq   = tid & 63;   // float4 index within F

  for (int rt = 0; rt < 2; ++rt) {
    float acc[4][3][4];
    #pragma unroll
    for (int j = 0; j < 4; ++j)
      #pragma unroll
      for (int k = 0; k < 3; ++k)
        #pragma unroll
        for (int e = 0; e < 4; ++e) acc[j][k][e] = 0.f;

    for (int y = 0; y < 17; ++y) {
      #pragma unroll
      for (int j = 0; j < 4; ++j) {
        const int    lrow = rt * 16 + j * 4 + rsub;
        const size_t rg   = (size_t)(rowBase + lrow);
        const float4* src = (y == 0)
            ? (const float4*)(nodes + rg * Fdim)
            : (const float4*)(cvec + (rg * Cdim + (size_t)(y - 1)) * Fdim);
        float4 v = src[fq];                       // coalesced 16B per lane
        float c0 = coef[lrow][y][0];
        float c1 = coef[lrow][y][1];
        float c2 = coef[lrow][y][2];
        acc[j][0][0] = fmaf(v.x, c0, acc[j][0][0]);
        acc[j][0][1] = fmaf(v.y, c0, acc[j][0][1]);
        acc[j][0][2] = fmaf(v.z, c0, acc[j][0][2]);
        acc[j][0][3] = fmaf(v.w, c0, acc[j][0][3]);
        acc[j][1][0] = fmaf(v.x, c1, acc[j][1][0]);
        acc[j][1][1] = fmaf(v.y, c1, acc[j][1][1]);
        acc[j][1][2] = fmaf(v.z, c1, acc[j][1][2]);
        acc[j][1][3] = fmaf(v.w, c1, acc[j][1][3]);
        acc[j][2][0] = fmaf(v.x, c2, acc[j][2][0]);
        acc[j][2][1] = fmaf(v.y, c2, acc[j][2][1]);
        acc[j][2][2] = fmaf(v.z, c2, acc[j][2][2]);
        acc[j][2][3] = fmaf(v.w, c2, acc[j][2][3]);
      }
    }
    // split to bf16 hi/lo and stash in LDS (A operand storage)
    #pragma unroll
    for (int j = 0; j < 4; ++j) {
      const int lrow = rt * 16 + j * 4 + rsub;
      #pragma unroll
      for (int k = 0; k < 3; ++k) {
        const int kkb = k * Fdim + fq * 4;
        #pragma unroll
        for (int e = 0; e < 4; ++e) {
          float  x = acc[j][k][e];
          bf16_t h = (bf16_t)x;
          Ah[lrow][kkb + e] = h;
          Al[lrow][kkb + e] = (bf16_t)(x - (float)h);
        }
      }
    }
  }
  __syncthreads();

  // ---- stage 2: WMMA GEMM [32 x 768] x [768 x 256] per WG ----
  const int wave    = tid >> 5;
  const int lane    = tid & 31;
  const int l16     = lane & 15;
  const int half    = lane >> 4;
  const int colBase = wave * 32;       // each wave owns 32 output columns
  const int koffA   = half * 8;
  const int koffB   = half * 8;

  v8f accum[2][2];
  #pragma unroll
  for (int rt = 0; rt < 2; ++rt)
    #pragma unroll
    for (int ct = 0; ct < 2; ++ct)
      #pragma unroll
      for (int r = 0; r < 8; ++r) accum[rt][ct][r] = 0.f;

  for (int step = 0; step < 24; ++step) {
    const int kk0 = step * 32;
    v16bf a0h = load_a16(&Ah[l16][kk0 + koffA]);
    v16bf a0l = load_a16(&Al[l16][kk0 + koffA]);
    v16bf a1h = load_a16(&Ah[16 + l16][kk0 + koffA]);
    v16bf a1l = load_a16(&Al[16 + l16][kk0 + koffA]);
    #pragma unroll
    for (int ct = 0; ct < 2; ++ct) {
      const int o = colBase + ct * 16 + l16;
      v16bf bh = load_b16(Wh + (size_t)o * Kdim + kk0 + koffB);
      v16bf bl = load_b16(Wl + (size_t)o * Kdim + kk0 + koffB);
      // fp32-split product: Ah*Bh + Al*Bh + Ah*Bl  (drop lo*lo)
      accum[0][ct] = __builtin_amdgcn_wmma_f32_16x16x32_bf16(
          false, a0h, false, bh, (short)0, accum[0][ct], false, false);
      accum[0][ct] = __builtin_amdgcn_wmma_f32_16x16x32_bf16(
          false, a0l, false, bh, (short)0, accum[0][ct], false, false);
      accum[0][ct] = __builtin_amdgcn_wmma_f32_16x16x32_bf16(
          false, a0h, false, bl, (short)0, accum[0][ct], false, false);
      accum[1][ct] = __builtin_amdgcn_wmma_f32_16x16x32_bf16(
          false, a1h, false, bh, (short)0, accum[1][ct], false, false);
      accum[1][ct] = __builtin_amdgcn_wmma_f32_16x16x32_bf16(
          false, a1l, false, bh, (short)0, accum[1][ct], false, false);
      accum[1][ct] = __builtin_amdgcn_wmma_f32_16x16x32_bf16(
          false, a1h, false, bl, (short)0, accum[1][ct], false, false);
    }
  }

  // ---- epilogue: bias + leaky_relu, C/D layout: lane=N, vgpr r -> M=r(+8) ----
  #pragma unroll
  for (int ct = 0; ct < 2; ++ct) {
    const int   col = colBase + ct * 16 + l16;
    const float cv  = convb[col];
    #pragma unroll
    for (int rt = 0; rt < 2; ++rt) {
      #pragma unroll
      for (int r = 0; r < 8; ++r) {
        const int row = rowBase + rt * 16 + half * 8 + r;
        float v = accum[rt][ct][r] + cv;
        out[(size_t)row * Odim + col] = (v > 0.f) ? v : 0.01f * v;
      }
    }
  }
}

// ---------------------------------------------------------------------------
extern "C" void kernel_launch(void* const* d_in, const int* in_sizes, int n_in,
                              void* d_out, int out_size, void* d_ws, size_t ws_size,
                              hipStream_t stream) {
  (void)in_sizes; (void)n_in; (void)out_size; (void)ws_size;
  const float* nodes    = (const float*)d_in[0];
  const float* cvec     = (const float*)d_in[1];
  const float* w_t      = (const float*)d_in[2];
  const float* w_l      = (const float*)d_in[3];
  const float* w_r      = (const float*)d_in[4];
  const float* convb    = (const float*)d_in[5];
  const int*   children = (const int*)d_in[6];

  bf16_t* Wh = (bf16_t*)d_ws;                         // 384 KB
  bf16_t* Wl = Wh + (size_t)3 * Fdim * Odim;          // +384 KB

  prep_weights_kernel<<<(3 * Fdim * Odim + 255) / 256, 256, 0, stream>>>(
      w_t, w_l, w_r, Wh, Wl);

  const int nRows = Bdim * Tdim;                      // 16384
  tree_conv_wmma_kernel<<<nRows / ROWS_PER_WG, 256, 0, stream>>>(
      nodes, cvec, convb, children, Wh, Wl, (float*)d_out);
}
